// MiMoAudioAttention_77738908057993
// MI455X (gfx1250) — matrix operand order repaired and checked by reading, not compiled, not run on hardware
//
#include <hip/hip_runtime.h>

// ---------------------------------------------------------------------------
// Types
// ---------------------------------------------------------------------------
typedef __attribute__((ext_vector_type(16))) __bf16       bf16x16;
typedef __attribute__((ext_vector_type(4)))  __bf16       bf16x4;
typedef __attribute__((ext_vector_type(8)))  float        f32x8;
typedef __attribute__((ext_vector_type(4)))  float        f32x4;
typedef __attribute__((ext_vector_type(4)))  unsigned int u32x4;
typedef __attribute__((ext_vector_type(4)))  int          i32x4;

#define B_  4
#define T_  2048
#define H_  2048
#define NH_ 16
#define NKV_ 4
#define HD_ 128
#define SCALE_ 0.08838834764831845f   // 128^-0.5
#define NEG_BIG -3.0e38f

#if defined(__has_builtin)
#if __has_builtin(__builtin_amdgcn_global_load_async_to_lds_b128)
#define ASYNC_LDS 1
#endif
#endif

struct Pair16 { u32x4 a, b; };

static __device__ __forceinline__ bf16x16 ldfrag(const __bf16* p0, const __bf16* p1) {
  Pair16 t;
  t.a = *(const u32x4*)p0;
  t.b = *(const u32x4*)p1;
  return __builtin_bit_cast(bf16x16, t);
}

static __device__ __forceinline__ f32x8 wmma_bf16(bf16x16 a, bf16x16 b, f32x8 c) {
  // (neg_a, A, neg_b, B, c_mod, C, reuse_a, reuse_b)
  return __builtin_amdgcn_wmma_f32_16x16x32_bf16(false, a, false, b, (short)0, c,
                                                 false, false);
}

// 16-byte global -> LDS copy: async path when available, else through VGPRs
static __device__ __forceinline__ void cp16(__bf16* lds, const __bf16* g) {
#ifdef ASYNC_LDS
  __builtin_amdgcn_global_load_async_to_lds_b128(
      (__attribute__((address_space(1))) i32x4*)g,
      (__attribute__((address_space(3))) i32x4*)lds, 0, 0);
#else
  *(u32x4*)lds = *(const u32x4*)g;
#endif
}

static __device__ __forceinline__ void wait_async() {
#ifdef ASYNC_LDS
  asm volatile("s_wait_asynccnt 0" ::: "memory");
#endif
}

// ---------------------------------------------------------------------------
// 1) Weight prep: f32 [K][N] -> bf16 transposed [N][K]  (LDS-tiled, coalesced)
// ---------------------------------------------------------------------------
__global__ __launch_bounds__(256) void k_wt_transpose(const float* __restrict__ src,
                                                      __bf16* __restrict__ dst,
                                                      int N, int K) {
  __shared__ float tile[32][33];
  const int tid = threadIdx.x;
  const int tx = tid & 31, ty = tid >> 5;
  const int n0 = blockIdx.x * 32, k0 = blockIdx.y * 32;
#pragma unroll
  for (int j = 0; j < 4; ++j) {
    int k = ty + j * 8;
    tile[k][tx] = src[(size_t)(k0 + k) * N + n0 + tx];
  }
  __syncthreads();
#pragma unroll
  for (int j = 0; j < 4; ++j) {
    int n = ty + j * 8;
    dst[(size_t)(n0 + n) * K + k0 + tx] = (__bf16)tile[tx][n];
  }
}

// ---------------------------------------------------------------------------
// 2) QKV projection GEMM: C[8192,N] = bf16(X f32)[8192,2048] @ Wt^T + bias
//    Wt is pre-transposed bf16 [N][2048]. Epilogue scatters bf16 with strides
//    (st_t, st_d) so V can be written transposed ([B,NKV,HD,T]) for free.
// ---------------------------------------------------------------------------
__global__ __launch_bounds__(256) void k_gemm_qkv(const float* __restrict__ A,
                                                  const __bf16* __restrict__ Wt,
                                                  const float* __restrict__ bias,
                                                  __bf16* __restrict__ out,
                                                  int nheads, int st_t, int st_d) {
  __shared__ __bf16 As[128 * 72];   // [m][k], stride 72
  __shared__ __bf16 Bs[128 * 72];   // [n][k], stride 72
  const int tid  = threadIdx.x;
  const int lane = tid & 31, wid = tid >> 5;
  const int wm = wid & 1, wn = wid >> 1;        // 2 x 4 wave grid -> 64x32 per wave
  const int g = lane >> 4, r = lane & 15;
  const int m0 = blockIdx.y * 128, n0 = blockIdx.x * 128;

  f32x8 acc[4][2];
#pragma unroll
  for (int mt = 0; mt < 4; ++mt)
#pragma unroll
    for (int nt = 0; nt < 2; ++nt)
#pragma unroll
      for (int e = 0; e < 8; ++e) acc[mt][nt][e] = 0.0f;

  for (int k0 = 0; k0 < 2048; k0 += 64) {
    // stage B: contiguous rows of Wt[n][k] (async to LDS, 1024 x b128)
#pragma unroll
    for (int j = 0; j < 4; ++j) {
      int idx8 = tid + j * 256;
      int n = idx8 >> 3, c8 = idx8 & 7;
      cp16(Bs + n * 72 + c8 * 8, Wt + (size_t)(n0 + n) * 2048 + k0 + c8 * 8);
    }
    // stage A: 128x64 f32 -> bf16 (needs VALU cvt, stays synchronous)
#pragma unroll
    for (int j = 0; j < 8; ++j) {
      int idx4 = tid + j * 256;
      int row = idx4 >> 4, c4 = idx4 & 15;
      f32x4 v = *(const f32x4*)(A + (size_t)(m0 + row) * 2048 + k0 + c4 * 4);
      bf16x4 o;
      o[0] = (__bf16)v[0]; o[1] = (__bf16)v[1]; o[2] = (__bf16)v[2]; o[3] = (__bf16)v[3];
      *(bf16x4*)(As + row * 72 + c4 * 4) = o;
    }
    wait_async();
    __syncthreads();

#pragma unroll
    for (int kk = 0; kk < 64; kk += 32) {
      bf16x16 af[4], bfg[2];
#pragma unroll
      for (int mt = 0; mt < 4; ++mt) {
        const __bf16* base = As + (wm * 64 + mt * 16 + r) * 72 + kk;
        af[mt] = ldfrag(base + g * 8, base + 16 + g * 8);
      }
#pragma unroll
      for (int nt = 0; nt < 2; ++nt) {
        const __bf16* base = Bs + (wn * 32 + nt * 16 + r) * 72 + kk;
        bfg[nt] = ldfrag(base + g * 8, base + 16 + g * 8);
      }
#pragma unroll
      for (int mt = 0; mt < 4; ++mt)
#pragma unroll
        for (int nt = 0; nt < 2; ++nt)
          acc[mt][nt] = wmma_bf16(af[mt], bfg[nt], acc[mt][nt]);
    }
    __syncthreads();
  }

  // epilogue: bias + strided scatter into per-head layout
#pragma unroll
  for (int mt = 0; mt < 4; ++mt)
#pragma unroll
    for (int nt = 0; nt < 2; ++nt) {
      int gc = n0 + wn * 32 + nt * 16 + r;
      float bv = bias[gc];
      int hh = gc >> 7, d = gc & 127;
#pragma unroll
      for (int i = 0; i < 8; ++i) {
        int gm = m0 + wm * 64 + mt * 16 + i + 8 * g;
        int b = gm >> 11, t = gm & 2047;
        float v = acc[mt][nt][i] + bv;
        out[(size_t)(b * nheads + hh) * (T_ * HD_) + (size_t)t * st_t +
            (size_t)d * st_d] = (__bf16)v;
      }
    }
}

// ---------------------------------------------------------------------------
// 3) RoPE in place on bf16 [B, nh, T, HD]
// ---------------------------------------------------------------------------
__global__ void k_rope(__bf16* __restrict__ buf, const int* __restrict__ positions,
                       int nheads, int total) {
  int i = blockIdx.x * blockDim.x + threadIdx.x;
  if (i >= total) return;
  int d = i & 63;
  int rest = i >> 6;            // (b*nh + h)*T + t
  int t = rest & (T_ - 1);
  int bh = rest >> 11;
  int b = bh / nheads;
  float pos = (float)positions[b * T_ + t];
  float inv = __expf(-(float)d * 0.14391156831212787f);  // ln(10000)/64
  float f = pos * inv, s, c;
  __sincosf(f, &s, &c);
  size_t base = (size_t)rest * HD_;
  float x1 = (float)buf[base + d], x2 = (float)buf[base + d + 64];
  buf[base + d]      = (__bf16)(x1 * c - x2 * s);
  buf[base + d + 64] = (__bf16)(x2 * c + x1 * s);
}

// ---------------------------------------------------------------------------
// 4) Causal flash attention with GQA. Block = 4 waves, 64 q rows per block.
//    K in [B,NKV,T,HD]; V pre-transposed in [B,NKV,HD,T].
//    Writes bf16 attention output in [B*T, NH*HD] row-major.
// ---------------------------------------------------------------------------
__global__ __launch_bounds__(128) void k_flash(const __bf16* __restrict__ Q,
                                               const __bf16* __restrict__ K,
                                               const __bf16* __restrict__ Vt,
                                               __bf16* __restrict__ O) {
  __shared__ __bf16 Ks[64 * 136];      // [j][d]
  __shared__ __bf16 Vs[128 * 72];      // [d][j]
  __shared__ __bf16 Ps[4][16 * 72];    // per-wave P re-layout patch
  const int tid = threadIdx.x;
  const int lane = tid & 31, w = tid >> 5;
  const int g = lane >> 4, r = lane & 15;
  const int qb = blockIdx.x, bh = blockIdx.y;
  const int b = bh >> 4, h = bh & 15, kv = h >> 2;

  const __bf16* Qp  = Q  + ((size_t)bh * T_ + qb * 64 + w * 16) * HD_;
  const __bf16* Kp  = K  + (size_t)(b * NKV_ + kv) * T_ * HD_;
  const __bf16* Vtp = Vt + (size_t)(b * NKV_ + kv) * T_ * HD_;   // [d][t]

  // Q as 4 A-fragments (16 rows x K=128)
  bf16x16 qf[4];
#pragma unroll
  for (int ks = 0; ks < 4; ++ks) {
    const __bf16* base = Qp + r * HD_ + ks * 32;
    qf[ks] = ldfrag(base + g * 8, base + 16 + g * 8);
  }

  f32x8 of[8];
  float m[8], l[8];
#pragma unroll
  for (int dt = 0; dt < 8; ++dt)
#pragma unroll
    for (int e = 0; e < 8; ++e) of[dt][e] = 0.0f;
#pragma unroll
  for (int i = 0; i < 8; ++i) { m[i] = NEG_BIG; l[i] = 0.0f; }

  const int q_base = qb * 64 + w * 16;
  const int njb = qb + 1;

  for (int jb = 0; jb < njb; ++jb) {
    const int j0 = jb * 64;
    // stage K block: [j][d], contiguous (1024 x b128, async)
#pragma unroll
    for (int j = 0; j < 8; ++j) {
      int idx8 = tid + j * 128;
      int row = idx8 >> 4, c8 = idx8 & 15;
      cp16(Ks + row * 136 + c8 * 8, Kp + (size_t)(j0 + row) * HD_ + c8 * 8);
    }
    // stage V block: rows of Vt[d][t], contiguous (1024 x b128, async)
#pragma unroll
    for (int j = 0; j < 8; ++j) {
      int idx8 = tid + j * 128;
      int d = idx8 >> 3, c8 = idx8 & 7;
      cp16(Vs + d * 72 + c8 * 8, Vtp + (size_t)d * T_ + j0 + c8 * 8);
    }
    wait_async();
    __syncthreads();

    // S = Q K^T (16 x 64)
    f32x8 sf[4];
#pragma unroll
    for (int jt = 0; jt < 4; ++jt)
#pragma unroll
      for (int e = 0; e < 8; ++e) sf[jt][e] = 0.0f;
#pragma unroll
    for (int ks = 0; ks < 4; ++ks)
#pragma unroll
      for (int jt = 0; jt < 4; ++jt) {
        const __bf16* base = Ks + (jt * 16 + r) * 136 + ks * 32;
        bf16x16 bb = ldfrag(base + g * 8, base + 16 + g * 8);
        sf[jt] = wmma_bf16(qf[ks], bb, sf[jt]);
      }

    // online softmax (row = i + 8*g, col = r within each 16-wide j tile)
#pragma unroll
    for (int i = 0; i < 8; ++i) {
      int q = q_base + i + 8 * g;
      float sv[4], pmax = NEG_BIG;
#pragma unroll
      for (int jt = 0; jt < 4; ++jt) {
        float x = sf[jt][i] * SCALE_;
        int j = j0 + jt * 16 + r;
        x = (j <= q) ? x : NEG_BIG;
        sv[jt] = x;
        pmax = fmaxf(pmax, x);
      }
#pragma unroll
      for (int off = 8; off > 0; off >>= 1)
        pmax = fmaxf(pmax, __shfl_xor(pmax, off));
      float mn = fmaxf(m[i], pmax);
      float alpha = __expf(m[i] - mn);
      float psum = 0.0f;
#pragma unroll
      for (int jt = 0; jt < 4; ++jt) {
        float p = __expf(sv[jt] - mn);
        psum += p;
        Ps[w][(i + 8 * g) * 72 + jt * 16 + r] = (__bf16)p;
      }
#pragma unroll
      for (int off = 8; off > 0; off >>= 1) psum += __shfl_xor(psum, off);
      l[i] = l[i] * alpha + psum;
      m[i] = mn;
#pragma unroll
      for (int dt = 0; dt < 8; ++dt) of[dt][i] *= alpha;
    }

    // O += P * V
#pragma unroll
    for (int jk = 0; jk < 64; jk += 32) {
      const __bf16* pbase = &Ps[w][r * 72 + jk];
      bf16x16 pa = ldfrag(pbase + g * 8, pbase + 16 + g * 8);
#pragma unroll
      for (int dt = 0; dt < 8; ++dt) {
        const __bf16* vbase = Vs + (dt * 16 + r) * 72 + jk;
        bf16x16 vb = ldfrag(vbase + g * 8, vbase + 16 + g * 8);
        of[dt] = wmma_bf16(pa, vb, of[dt]);
      }
    }
    __syncthreads();
  }

  // normalize and write [B*T, NH*HD] bf16
#pragma unroll
  for (int i = 0; i < 8; ++i) {
    int q = q_base + i + 8 * g;
    float inv = 1.0f / l[i];
    size_t orow = ((size_t)b * T_ + q) * (NH_ * HD_) + h * HD_;
#pragma unroll
    for (int dt = 0; dt < 8; ++dt)
      O[orow + dt * 16 + r] = (__bf16)(of[dt][i] * inv);
  }
}

// ---------------------------------------------------------------------------
// 5) Output GEMM: out f32[8192,2048] = Ob bf16[8192,2048] @ Wot^T
//    (Wot pre-transposed bf16 [2048][2048]); fully async staging.
// ---------------------------------------------------------------------------
__global__ __launch_bounds__(256) void k_gemm_out(const __bf16* __restrict__ A,
                                                  const __bf16* __restrict__ Wt,
                                                  float* __restrict__ out) {
  __shared__ __bf16 As[128 * 72];
  __shared__ __bf16 Bs[128 * 72];
  const int tid = threadIdx.x;
  const int lane = tid & 31, wid = tid >> 5;
  const int wm = wid & 1, wn = wid >> 1;
  const int g = lane >> 4, r = lane & 15;
  const int m0 = blockIdx.y * 128, n0 = blockIdx.x * 128;

  f32x8 acc[4][2];
#pragma unroll
  for (int mt = 0; mt < 4; ++mt)
#pragma unroll
    for (int nt = 0; nt < 2; ++nt)
#pragma unroll
      for (int e = 0; e < 8; ++e) acc[mt][nt][e] = 0.0f;

  for (int k0 = 0; k0 < 2048; k0 += 64) {
#pragma unroll
    for (int j = 0; j < 4; ++j) {  // A rows, contiguous
      int idx8 = tid + j * 256;
      int row = idx8 >> 3, c8 = idx8 & 7;
      cp16(As + row * 72 + c8 * 8, A + (size_t)(m0 + row) * 2048 + k0 + c8 * 8);
    }
#pragma unroll
    for (int j = 0; j < 4; ++j) {  // B rows of Wt[n][k], contiguous
      int idx8 = tid + j * 256;
      int n = idx8 >> 3, c8 = idx8 & 7;
      cp16(Bs + n * 72 + c8 * 8, Wt + (size_t)(n0 + n) * 2048 + k0 + c8 * 8);
    }
    wait_async();
    __syncthreads();
#pragma unroll
    for (int kk = 0; kk < 64; kk += 32) {
      bf16x16 af[4], bfg[2];
#pragma unroll
      for (int mt = 0; mt < 4; ++mt) {
        const __bf16* base = As + (wm * 64 + mt * 16 + r) * 72 + kk;
        af[mt] = ldfrag(base + g * 8, base + 16 + g * 8);
      }
#pragma unroll
      for (int nt = 0; nt < 2; ++nt) {
        const __bf16* base = Bs + (wn * 32 + nt * 16 + r) * 72 + kk;
        bfg[nt] = ldfrag(base + g * 8, base + 16 + g * 8);
      }
#pragma unroll
      for (int mt = 0; mt < 4; ++mt)
#pragma unroll
        for (int nt = 0; nt < 2; ++nt)
          acc[mt][nt] = wmma_bf16(af[mt], bfg[nt], acc[mt][nt]);
    }
    __syncthreads();
  }

#pragma unroll
  for (int mt = 0; mt < 4; ++mt)
#pragma unroll
    for (int nt = 0; nt < 2; ++nt) {
      int gc = n0 + wn * 32 + nt * 16 + r;
#pragma unroll
      for (int i = 0; i < 8; ++i) {
        int gm = m0 + wm * 64 + mt * 16 + i + 8 * g;
        out[(size_t)gm * 2048 + gc] = acc[mt][nt][i];
      }
    }
}

// ---------------------------------------------------------------------------
// Launch
// ---------------------------------------------------------------------------
extern "C" void kernel_launch(void* const* d_in, const int* in_sizes, int n_in,
                              void* d_out, int out_size, void* d_ws, size_t ws_size,
                              hipStream_t stream) {
  const float* hs  = (const float*)d_in[0];
  const int*   pos = (const int*)d_in[1];
  const float* wq  = (const float*)d_in[2];
  const float* bq  = (const float*)d_in[3];
  const float* wk  = (const float*)d_in[4];
  const float* bk  = (const float*)d_in[5];
  const float* wv  = (const float*)d_in[6];
  const float* bv  = (const float*)d_in[7];
  const float* wo  = (const float*)d_in[8];
  float* out = (float*)d_out;

  char* ws = (char*)d_ws;
  size_t off = 0;
  auto take = [&](size_t bytes) -> char* {
    char* p = ws + off;
    off += (bytes + 255) & ~(size_t)255;
    return p;
  };
  __bf16* Wqt = (__bf16*)take((size_t)2048 * 2048 * 2);  // [N][K] bf16
  __bf16* Wkt = (__bf16*)take((size_t)512 * 2048 * 2);
  __bf16* Wvt = (__bf16*)take((size_t)512 * 2048 * 2);
  __bf16* Wot = (__bf16*)take((size_t)2048 * 2048 * 2);
  __bf16* Qb  = (__bf16*)take((size_t)B_ * NH_ * T_ * HD_ * 2);   // [B,NH,T,HD]
  __bf16* Kb  = (__bf16*)take((size_t)B_ * NKV_ * T_ * HD_ * 2);  // [B,NKV,T,HD]
  __bf16* Vtb = (__bf16*)take((size_t)B_ * NKV_ * T_ * HD_ * 2);  // [B,NKV,HD,T]
  __bf16* Ob  = (__bf16*)take((size_t)8192 * 2048 * 2);           // [B*T, NH*HD]

  // weight convert + transpose (f32 [K][N] -> bf16 [N][K])
  k_wt_transpose<<<dim3(2048 / 32, 2048 / 32), 256, 0, stream>>>(wq, Wqt, 2048, 2048);
  k_wt_transpose<<<dim3(512 / 32, 2048 / 32), 256, 0, stream>>>(wk, Wkt, 512, 2048);
  k_wt_transpose<<<dim3(512 / 32, 2048 / 32), 256, 0, stream>>>(wv, Wvt, 512, 2048);
  k_wt_transpose<<<dim3(2048 / 32, 2048 / 32), 256, 0, stream>>>(wo, Wot, 2048, 2048);

  // QKV projections (V written transposed: st_t=1, st_d=T)
  k_gemm_qkv<<<dim3(16, 64), 256, 0, stream>>>(hs, Wqt, bq, Qb, NH_, HD_, 1);
  k_gemm_qkv<<<dim3(4, 64), 256, 0, stream>>>(hs, Wkt, bk, Kb, NKV_, HD_, 1);
  k_gemm_qkv<<<dim3(4, 64), 256, 0, stream>>>(hs, Wvt, bv, Vtb, NKV_, 1, T_);

  // RoPE on Q and K
  int nq = B_ * NH_ * T_ * 64;
  int nk = B_ * NKV_ * T_ * 64;
  k_rope<<<(nq + 255) / 256, 256, 0, stream>>>(Qb, pos, NH_, nq);
  k_rope<<<(nk + 255) / 256, 256, 0, stream>>>(Kb, pos, NKV_, nk);

  // flash attention
  k_flash<<<dim3(T_ / 64, B_ * NH_), 128, 0, stream>>>(Qb, Kb, Vtb, Ob);

  // output projection
  k_gemm_out<<<dim3(16, 64), 256, 0, stream>>>(Ob, Wot, out);
}
